// StrGNNLayer_43920335569332
// MI455X (gfx1250) — matrix operand order repaired and checked by reading, not compiled, not run on hardware
//
#include <hip/hip_runtime.h>
#include <math.h>

// ---------------------------------------------------------------------------
// Types for CDNA5 WMMA (wave32): V_WMMA_F32_16X16X32_F16
// ---------------------------------------------------------------------------
typedef __attribute__((ext_vector_type(16))) _Float16 v16h;
typedef __attribute__((ext_vector_type(8)))  _Float16 v8h;
typedef __attribute__((ext_vector_type(8)))  float    v8f;

#define Hh   1024
#define Bb   16
#define Tt   2048
#define Ss   64
#define Ll   3
#define ROWS (Bb * Ss)          // 1024  (M dimension of the big GEMMs)
#define CH   (Ll * Hh)          // 3072  (concat width / final K)

// ---------------------------------------------------------------------------
// WMMA fragment loaders (layouts per CDNA5 ISA §7.12.2)
// A (16x32 f16, row-major source, lda = K):
//   lane L: M = L%16 ; halves 0..7 -> K = kb..kb+7 ; halves 8..15 -> K = kb+16..kb+23
//   where kb = k0 + (L&16 ? 8 : 0)
// B from N-major copy BT[N,K] (ldb = K):
//   lane L: N = L%16 ; same K mapping as A (kb..kb+7, kb+16..kb+23)
// ---------------------------------------------------------------------------
__device__ __forceinline__ v16h load_a_frag(const _Float16* A, int lda, int m,
                                            int k0, int lane) {
  int kb = k0 + ((lane & 16) ? 8 : 0);
  const _Float16* p = A + (size_t)m * lda + kb;
  v8h lo = *(const v8h*)(p);
  v8h hi = *(const v8h*)(p + 16);
  v16h r;
#pragma unroll
  for (int i = 0; i < 8; ++i) { r[i] = lo[i]; r[i + 8] = hi[i]; }
  return r;
}

__device__ __forceinline__ v16h load_b_frag(const _Float16* BT, int ldb, int n,
                                            int k0, int lane) {
  int kb = k0 + ((lane & 16) ? 8 : 0);
  const _Float16* p = BT + (size_t)n * ldb + kb;
  v8h lo = *(const v8h*)(p);
  v8h hi = *(const v8h*)(p + 16);
  v16h r;
#pragma unroll
  for (int i = 0; i < 8; ++i) { r[i] = lo[i]; r[i + 8] = hi[i]; }
  return r;
}

// One wave computes a 16x64 strip: 4 accumulators, A fragment reused 4x.
__device__ __forceinline__ void gemm_acc4(const _Float16* A, int lda,
                                          const _Float16* BT, int ldb, int K,
                                          int row0, int col0, int lane,
                                          v8f acc[4]) {
  int am = row0 + (lane & 15);
  for (int k0 = 0; k0 < K; k0 += 32) {
    v16h a = load_a_frag(A, lda, am, k0, lane);
#pragma unroll
    for (int j = 0; j < 4; ++j) {
      v16h b = load_b_frag(BT, ldb, col0 + 16 * j + (lane & 15), k0, lane);
      acc[j] = __builtin_amdgcn_wmma_f32_16x16x32_f16(
          false, a, false, b, (short)0, acc[j], false, false);
    }
  }
}

// ---------------------------------------------------------------------------
// Stage 1: segment start extraction, pooled sentence means (HBM-bound pass)
// ---------------------------------------------------------------------------
__global__ void k_seg_starts(const int* __restrict__ segments, int* __restrict__ starts) {
  int b = blockIdx.x;
  if (threadIdx.x != 0) return;
  const int* row = segments + (size_t)b * Tt;
  int idx = 0;
  for (int t = 0; t < Tt; ++t)
    if (row[t]) { if (idx < Ss) starts[b * (Ss + 1) + idx] = t; ++idx; }
  starts[b * (Ss + 1) + Ss] = Tt - 1;   // torch quirk: last token excluded
}

__global__ void k_seg_mean(const float* __restrict__ x, const int* __restrict__ starts,
                           float* __restrict__ sent) {
  int blk = blockIdx.x;              // b*Ss + s
  int b = blk >> 6, s = blk & 63;
  int st = starts[b * (Ss + 1) + s];
  int en = starts[b * (Ss + 1) + s + 1];
  int tid = threadIdx.x;             // 256 threads, 4 columns each
  float s0 = 0.f, s1 = 0.f, s2 = 0.f, s3 = 0.f;
  for (int t = st; t < en; ++t) {
    const float* row = x + ((size_t)t * Bb + b) * Hh;
    s0 += row[tid];       s1 += row[tid + 256];
    s2 += row[tid + 512]; s3 += row[tid + 768];
  }
  float inv = 1.f / (float)((en - st) > 0 ? (en - st) : 1);
  float* o = sent + (size_t)blk * Hh;
  o[tid] = s0 * inv; o[tid + 256] = s1 * inv;
  o[tid + 512] = s2 * inv; o[tid + 768] = s3 * inv;
}

__global__ void k_mask_pad(const float* __restrict__ sent, float* __restrict__ mask,
                           float* __restrict__ pad_out) {
  int i = blockIdx.x * blockDim.x + threadIdx.x;
  if (i >= ROWS) return;
  bool real = fabsf(sent[(size_t)i * Hh]) > 0.f;
  mask[i] = real ? 1.f : 0.f;
  pad_out[i] = real ? 0.f : 1.f;     // ~real_mask
}

__global__ void k_f32_to_f16(const float* __restrict__ src, _Float16* __restrict__ dst, int n) {
  int i = blockIdx.x * blockDim.x + threadIdx.x;
  if (i < n) dst[i] = (_Float16)src[i];
}

// W [K,N] f32  ->  WT [N,K] f16 (coalesced reads)
__global__ void k_transpose_w(const float* __restrict__ W, _Float16* __restrict__ WT,
                              int K, int N) {
  int idx = blockIdx.x * blockDim.x + threadIdx.x;
  if (idx >= K * N) return;
  int k = idx / N, n = idx - k * N;
  WT[(size_t)n * K + k] = (_Float16)W[idx];
}

// h [B*S,H] f32 -> hT [B,H,S] f16 (per-batch N-major copy for the ctx GEMM)
__global__ void k_transpose_h(const float* __restrict__ h, _Float16* __restrict__ hT) {
  int idx = blockIdx.x * blockDim.x + threadIdx.x;   // B*S*H
  if (idx >= ROWS * Hh) return;
  int hh = idx & (Hh - 1);
  int row = idx >> 10;                 // b*Ss + s
  int b = row >> 6, s = row & 63;
  hT[((size_t)b * Hh + hh) * Ss + s] = (_Float16)h[idx];
}

// ---------------------------------------------------------------------------
// GEMM kernels (8 waves / 256 threads, scores 4 waves / 128; wave strip = 16x64)
// ---------------------------------------------------------------------------
// q/k: Out = f16(tanh(A @ B)), M=N=K=1024
__global__ void k_gemm_tanh(const _Float16* __restrict__ A, const _Float16* __restrict__ BT,
                            _Float16* __restrict__ Out) {
  int lane = threadIdx.x & 31, wave = threadIdx.x >> 5;
  int id = blockIdx.x * 8 + wave;              // 1024 wave-tiles
  int row0 = (id & 63) * 16, col0 = (id >> 6) * 64;
  v8f acc[4] = {};
  gemm_acc4(A, Hh, BT, Hh, Hh, row0, col0, lane, acc);
  int mb = row0 + ((lane & 16) ? 8 : 0), n0 = lane & 15;
#pragma unroll
  for (int j = 0; j < 4; ++j)
#pragma unroll
    for (int r = 0; r < 8; ++r)
      Out[(size_t)(mb + r) * Hh + col0 + 16 * j + n0] = (_Float16)tanhf(acc[j][r]);
}

// scores[b] = (q[b] @ k[b]^T) / sqrt(H)   (k16 is already N-major for this)
__global__ void k_gemm_scores(const _Float16* __restrict__ q16,
                              const _Float16* __restrict__ k16,
                              float* __restrict__ scores) {
  int lane = threadIdx.x & 31, wave = threadIdx.x >> 5;   // block = 128 (4 waves)
  int b = blockIdx.x;
  const _Float16* A  = q16 + (size_t)b * Ss * Hh;
  const _Float16* BT = k16 + (size_t)b * Ss * Hh;
  int row0 = wave * 16;
  v8f acc[4] = {};
  gemm_acc4(A, Hh, BT, Hh, Hh, row0, 0, lane, acc);
  float* o = scores + (size_t)b * Ss * Ss;
  int mb = row0 + ((lane & 16) ? 8 : 0), n0 = lane & 15;
#pragma unroll
  for (int j = 0; j < 4; ++j)
#pragma unroll
    for (int r = 0; r < 8; ++r)
      o[(size_t)(mb + r) * Ss + 16 * j + n0] = acc[j][r] * 0.03125f; // 1/sqrt(1024)
}

// ctx[b] = int_m[b] @ h[b]   (M=64, K=64, N=1024), A=intm16, BT=hT16[b]
__global__ void k_gemm_ctx(const _Float16* __restrict__ intm16,
                           const _Float16* __restrict__ hT16,
                           _Float16* __restrict__ ctx16) {
  int lane = threadIdx.x & 31, wave = threadIdx.x >> 5;
  int id = blockIdx.x * 8 + wave;     // 16 batches * 64 wave-tiles
  int b = id >> 6, t = id & 63;
  int row0 = (t & 3) * 16, col0 = (t >> 2) * 64;
  const _Float16* A  = intm16 + (size_t)b * Ss * Ss;
  const _Float16* BT = hT16  + (size_t)b * Hh * Ss;
  v8f acc[4] = {};
  gemm_acc4(A, Ss, BT, Ss, Ss, row0, col0, lane, acc);
  _Float16* o = ctx16 + (size_t)b * Ss * Hh;
  int mb = row0 + ((lane & 16) ? 8 : 0), n0 = lane & 15;
#pragma unroll
  for (int j = 0; j < 4; ++j)
#pragma unroll
    for (int r = 0; r < 8; ++r)
      o[(size_t)(mb + r) * Hh + col0 + 16 * j + n0] = (_Float16)acc[j][r];
}

// h_new = (relu(ctx @ Wo) + h_old) * mask ; also writes f16 copy + concat slice
__global__ void k_gemm_relu_res(const _Float16* __restrict__ A, const _Float16* __restrict__ BT,
                                const float* __restrict__ h_old, const float* __restrict__ mask,
                                float* __restrict__ h_new, _Float16* __restrict__ h16,
                                _Float16* __restrict__ concat, int layer) {
  int lane = threadIdx.x & 31, wave = threadIdx.x >> 5;
  int id = blockIdx.x * 8 + wave;
  int row0 = (id & 63) * 16, col0 = (id >> 6) * 64;
  v8f acc[4] = {};
  gemm_acc4(A, Hh, BT, Hh, Hh, row0, col0, lane, acc);
  int mb = row0 + ((lane & 16) ? 8 : 0), n0 = lane & 15;
#pragma unroll
  for (int j = 0; j < 4; ++j)
#pragma unroll
    for (int r = 0; r < 8; ++r) {
      int m = mb + r, n = col0 + 16 * j + n0;
      float v = fmaxf(acc[j][r], 0.f);
      v = (v + h_old[(size_t)m * Hh + n]) * mask[m];
      h_new[(size_t)m * Hh + n] = v;
      h16[(size_t)m * Hh + n] = (_Float16)v;
      concat[(size_t)m * CH + layer * Hh + n] = (_Float16)v;
    }
}

// final: out = fp16((relu(concat @ Wf + bf) + res) * mask), transposed to [S,B,H]
__global__ void k_gemm_final(const _Float16* __restrict__ A, const _Float16* __restrict__ BT,
                             const float* __restrict__ bf, const float* __restrict__ res,
                             const float* __restrict__ mask, float* __restrict__ out) {
  int lane = threadIdx.x & 31, wave = threadIdx.x >> 5;
  int id = blockIdx.x * 8 + wave;
  int row0 = (id & 63) * 16, col0 = (id >> 6) * 64;
  v8f acc[4] = {};
  gemm_acc4(A, CH, BT, CH, CH, row0, col0, lane, acc);
  int mb = row0 + ((lane & 16) ? 8 : 0), n0 = lane & 15;
#pragma unroll
  for (int j = 0; j < 4; ++j)
#pragma unroll
    for (int r = 0; r < 8; ++r) {
      int m = mb + r, n = col0 + 16 * j + n0;
      float v = fmaxf(acc[j][r] + bf[n], 0.f);
      v = (v + res[(size_t)m * Hh + n]) * mask[m];
      int b = m >> 6, s = m & 63;
      out[((size_t)s * Bb + b) * Hh + n] = (float)(_Float16)v;  // .half() rounding
    }
}

// ---------------------------------------------------------------------------
// Softmax / r_prob / sent_lens helper kernels
// ---------------------------------------------------------------------------
__global__ void k_softmax_intm(const float* __restrict__ scores, const float* __restrict__ mask,
                               float* __restrict__ out_intm, _Float16* __restrict__ intm16,
                               int layer) {
  int row = blockIdx.x;            // b*Ss + s
  int b = row >> 6, s = row & 63;
  int t = threadIdx.x;             // 64 threads
  __shared__ float sh[Ss];
  float v = (mask[b * Ss + t] > 0.f) ? scores[(size_t)row * Ss + t] : -1e9f;
  sh[t] = v; __syncthreads();
  for (int off = 32; off > 0; off >>= 1) {
    if (t < off) sh[t] = fmaxf(sh[t], sh[t + off]);
    __syncthreads();
  }
  float mx = sh[0]; __syncthreads();
  float e = __expf(v - mx);
  sh[t] = e; __syncthreads();
  for (int off = 32; off > 0; off >>= 1) {
    if (t < off) sh[t] += sh[t + off];
    __syncthreads();
  }
  float p = e / sh[0];
  out_intm[(((size_t)b * Ll + layer) * Ss + s) * Ss + t] = p;
  intm16[(size_t)row * Ss + t] = (_Float16)p;
}

__global__ void k_rlogits(const float* __restrict__ h, const float* __restrict__ Wr,
                          float* __restrict__ logits) {
  int lane = threadIdx.x & 31;
  int row = blockIdx.x * 8 + (threadIdx.x >> 5);   // 1024 rows
  const float* hr = h + (size_t)row * Hh;
  float p = 0.f;
  for (int i = lane; i < Hh; i += 32) p += hr[i] * Wr[i];
  for (int off = 16; off > 0; off >>= 1) p += __shfl_xor(p, off, 32);
  if (lane == 0) logits[row] = p;
}

__global__ void k_rprob_softmax(const float* __restrict__ logits, const float* __restrict__ mask,
                                float* __restrict__ out, int layer) {
  int b = blockIdx.x, t = threadIdx.x;   // 64 threads
  __shared__ float sh[Ss];
  float v = (mask[b * Ss + t] > 0.f) ? logits[b * Ss + t] : -1e9f;
  sh[t] = v; __syncthreads();
  for (int off = 32; off > 0; off >>= 1) {
    if (t < off) sh[t] = fmaxf(sh[t], sh[t + off]);
    __syncthreads();
  }
  float mx = sh[0]; __syncthreads();
  float e = __expf(v - mx);
  sh[t] = e; __syncthreads();
  for (int off = 32; off > 0; off >>= 1) {
    if (t < off) sh[t] += sh[t + off];
    __syncthreads();
  }
  out[((size_t)b * Ll + layer) * Ss + t] = e / sh[0];
}

__global__ void k_sent_lens(const int* __restrict__ segments, float* __restrict__ out) {
  int b = blockIdx.x, lane = threadIdx.x;   // 32 threads
  const int* row = segments + (size_t)b * Tt;
  int p = 0;
  for (int i = lane; i < Tt; i += 32) p += row[i];
  for (int off = 16; off > 0; off >>= 1) p += __shfl_xor(p, off, 32);
  if (lane == 0) out[b] = (float)p;
}

// ---------------------------------------------------------------------------
// Host launcher
// ---------------------------------------------------------------------------
extern "C" void kernel_launch(void* const* d_in, const int* in_sizes, int n_in,
                              void* d_out, int out_size, void* d_ws, size_t ws_size,
                              hipStream_t stream) {
  const float* x        = (const float*)d_in[0];
  const int*   segments = (const int*)d_in[2];
  const float* Wq       = (const float*)d_in[3];
  const float* Wk       = (const float*)d_in[4];
  const float* Wr       = (const float*)d_in[5];
  const float* Wo       = (const float*)d_in[6];
  const float* Wf       = (const float*)d_in[7];
  const float* bf       = (const float*)d_in[8];
  float* out = (float*)d_out;

  // carve workspace (256B aligned chunks)
  char* p = (char*)d_ws;
  auto carve = [&](size_t bytes) -> void* {
    void* r = (void*)p;
    p += (bytes + 255) & ~(size_t)255;
    return r;
  };
  int*       starts = (int*)carve(Bb * (Ss + 1) * sizeof(int));
  float*     sent   = (float*)carve((size_t)ROWS * Hh * 4);      // residual, kept
  float*     hA     = (float*)carve((size_t)ROWS * Hh * 4);
  float*     hB     = (float*)carve((size_t)ROWS * Hh * 4);
  _Float16*  h16    = (_Float16*)carve((size_t)ROWS * Hh * 2);
  _Float16*  q16    = (_Float16*)carve((size_t)ROWS * Hh * 2);
  _Float16*  k16    = (_Float16*)carve((size_t)ROWS * Hh * 2);
  _Float16*  ctx16  = (_Float16*)carve((size_t)ROWS * Hh * 2);
  _Float16*  hT16   = (_Float16*)carve((size_t)Bb * Hh * Ss * 2);
  _Float16*  WTq    = (_Float16*)carve((size_t)Hh * Hh * 2);
  _Float16*  WTk    = (_Float16*)carve((size_t)Hh * Hh * 2);
  _Float16*  WTo    = (_Float16*)carve((size_t)Hh * Hh * 2);
  _Float16*  WfT    = (_Float16*)carve((size_t)Hh * CH * 2);
  _Float16*  concat = (_Float16*)carve((size_t)ROWS * CH * 2);
  float*     scores = (float*)carve((size_t)Bb * Ss * Ss * 4);
  _Float16*  intm16 = (_Float16*)carve((size_t)Bb * Ss * Ss * 2);
  float*     mask   = (float*)carve(ROWS * 4);
  float*     rlog   = (float*)carve(ROWS * 4);

  // d_out flat layout (return order)
  float* out_gnn   = out;                                  // S*B*H = 1048576
  float* out_intm  = out + 1048576;                        // B*L*S*S = 196608
  float* out_rprob = out + 1048576 + 196608;               // B*L*S = 3072
  float* out_pad   = out + 1048576 + 196608 + 3072;        // B*S = 1024
  float* out_lens  = out + 1048576 + 196608 + 3072 + 1024; // B = 16

  // Stage 1: pooling (single HBM-bound pass over x) + masks + lens
  k_seg_starts<<<Bb, 32, 0, stream>>>(segments, starts);
  k_seg_mean<<<ROWS, 256, 0, stream>>>(x, starts, sent);
  k_mask_pad<<<4, 256, 0, stream>>>(sent, mask, out_pad);
  k_f32_to_f16<<<(ROWS * Hh) / 256, 256, 0, stream>>>(sent, h16, ROWS * Hh);
  k_transpose_w<<<(CH * Hh + 255) / 256, 256, 0, stream>>>(Wf, WfT, CH, Hh);
  k_sent_lens<<<Bb, 32, 0, stream>>>(segments, out_lens);

  // Stage 2: three HierGNN layers (all matmuls in f16 WMMA, f32 accumulate)
  const float* hcur = sent;
  float* hnext = hA;
  for (int l = 0; l < Ll; ++l) {
    const float* Wql = Wq + (size_t)l * Hh * Hh;
    const float* Wkl = Wk + (size_t)l * Hh * Hh;
    const float* Wol = Wo + (size_t)l * Hh * Hh;
    k_transpose_w<<<(Hh * Hh + 255) / 256, 256, 0, stream>>>(Wql, WTq, Hh, Hh);
    k_transpose_w<<<(Hh * Hh + 255) / 256, 256, 0, stream>>>(Wkl, WTk, Hh, Hh);
    k_transpose_w<<<(Hh * Hh + 255) / 256, 256, 0, stream>>>(Wol, WTo, Hh, Hh);

    k_gemm_tanh<<<128, 256, 0, stream>>>(h16, WTq, q16);
    k_gemm_tanh<<<128, 256, 0, stream>>>(h16, WTk, k16);
    k_gemm_scores<<<Bb, 128, 0, stream>>>(q16, k16, scores);
    k_softmax_intm<<<ROWS, Ss, 0, stream>>>(scores, mask, out_intm, intm16, l);

    k_transpose_h<<<(ROWS * Hh) / 256, 256, 0, stream>>>(hcur, hT16);
    k_gemm_ctx<<<128, 256, 0, stream>>>(intm16, hT16, ctx16);

    k_rlogits<<<128, 256, 0, stream>>>(hcur, Wr + (size_t)l * Hh, rlog);
    k_rprob_softmax<<<Bb, Ss, 0, stream>>>(rlog, mask, out_rprob, l);

    k_gemm_relu_res<<<128, 256, 0, stream>>>(ctx16, WTo, hcur, mask,
                                             hnext, h16, concat, l);
    hcur = hnext;
    hnext = (hcur == hA) ? hB : hA;
  }

  // Stage 3: final fused GEMM (K=3072) + bias + relu + residual + mask + fp16 + transpose
  k_gemm_final<<<128, 256, 0, stream>>>(concat, WfT, bf, sent, mask, out_gnn);
}
